// MaSLEncoder_57887569215624
// MI455X (gfx1250) — matrix operand order repaired and checked by reading, not compiled, not run on hardware
//
#include <hip/hip_runtime.h>
#include <hip/hip_bf16.h>
#include <stdint.h>

// ---------------------------------------------------------------------------
// Problem constants (from the reference)
// ---------------------------------------------------------------------------
#define N_FFT   200
#define HOP     100
#define N_FREQ  101     // N_FFT/2 + 1
#define KF_PAD  128     // N_FREQ padded to multiple of 32 for WMMA
#define EMB     256
#define D_INNER 512
#define D_STATE 128
#define D_CONV  4
#define DT_RANK 16
#define DT_PAD  32      // DT_RANK padded for WMMA
#define NCH     16      // channels in x
#define NFRAMES 39      // 1 + (4000-200)/100
#define BATCH   2
#define LSEQ    (NCH * NFRAMES)   // 624 (per batch)
#define ROWS    (BATCH * LSEQ)    // 1248 total sequence rows
#define DBL     (DT_RANK + 2 * D_STATE)  // 272
#define NW      17      // number of weight matrices to stage (1 proj + 4*4)

typedef __attribute__((ext_vector_type(16))) _Float16 v16h;
typedef __attribute__((ext_vector_type(8)))  float    v8f;

// ---------------------------------------------------------------------------
// STFT magnitude: direct real DFT, one thread per (frame, padded-bin).
// Emits the K-padded f16 matrix consumed by the projection WMMA GEMM.
// ---------------------------------------------------------------------------
__global__ void stft_kernel(const float* __restrict__ x, _Float16* __restrict__ spec_h) {
    int i = blockIdx.x * blockDim.x + threadIdx.x;
    if (i >= ROWS * KF_PAD) return;
    int r = i / KF_PAD;
    int k = i - r * KF_PAD;
    if (k >= N_FREQ) { spec_h[i] = (_Float16)0.0f; return; }
    int b  = r / LSEQ;
    int rc = r - b * LSEQ;
    int c  = rc / NFRAMES;
    int fr = rc - c * NFRAMES;
    const float* sig = x + ((size_t)b * NCH + c) * 4000 + (size_t)fr * HOP;
    float w = -2.0f * 3.14159265358979323846f * (float)k / (float)N_FFT;
    float re = 0.f, im = 0.f;
    for (int t = 0; t < N_FFT; ++t) {
        float s, co;
        __sincosf(w * (float)t, &s, &co);
        float sv = sig[t];
        re = __fmaf_rn(sv, co, re);
        im = __fmaf_rn(sv, s,  im);
    }
    spec_h[i] = (_Float16)sqrtf(re * re + im * im);
}

// ---------------------------------------------------------------------------
// Batched f32 -> f16 weight staging (all 17 matrices in ONE launch).
// Job table passed by value through kernarg.
// ---------------------------------------------------------------------------
struct CastJobs {
    const float* src[NW];
    _Float16*    dst[NW];
    int srcStride[NW];
    int K[NW];
    int Kpad[NW];
    int off[NW + 1];   // element prefix offsets into the combined iteration space
};

__global__ void cast_weights_kernel(CastJobs J) {
    int i = blockIdx.x * blockDim.x + threadIdx.x;
    if (i >= J.off[NW]) return;
    int j = 0;
    #pragma unroll
    for (int t = 0; t < NW; ++t)
        if (i >= J.off[t + 1]) j = t + 1;
    int e = i - J.off[j];
    int Kp = J.Kpad[j];
    int r = e / Kp;
    int k = e - r * Kp;
    J.dst[j][e] = (k < J.K[j]) ? (_Float16)J.src[j][(size_t)r * J.srcStride[j] + k]
                               : (_Float16)0.0f;
}

// ---------------------------------------------------------------------------
// WMMA GEMM: out[M,N] = A[M,Kpad] * W[N,Kpad]^T (+bias) (+softplus)
// Wave32 fragment layouts per CDNA5 ISA section 7.12.2:
//   A 16x32 f16 : lane&15 = row M; lane>>4 selects K-half; per lane 16 halfs
//                 covering K in {kh*8..kh*8+7, 16+kh*8..16+kh*8+7}
//   B 32x16 f16 : lane&15 = col N; same K order as A: {kh*8..kh*8+7, 16+kh*8..16+kh*8+7}
//   C/D 16x16 f32: lane&15 = col N; VGPR v holds row M = (lane>>4)*8 + v
// Optional fused f16 output (out_h, row-stride hKpad, valid cols < hK,
// zero-fill for hK <= col < hKpad) chains the next layer's WMMA input with
// no extra cast launch.
// ---------------------------------------------------------------------------
#define EP_NONE          0
#define EP_BIAS          1
#define EP_BIAS_SOFTPLUS 2

__device__ __forceinline__ v16h load_fragA(const _Float16* __restrict__ A,
                                           int row, int Kpad, int kbase, int khalf) {
    union { uint4 u[2]; v16h h; } f;
    const _Float16* p = A + (size_t)row * Kpad + kbase + khalf * 8;
    f.u[0] = *(const uint4*)(p);
    f.u[1] = *(const uint4*)(p + 16);
    return f.h;
}

__device__ __forceinline__ v16h load_fragB(const _Float16* __restrict__ W,
                                           int col, int Kpad, int kbase, int khalf) {
    union { uint4 u[2]; v16h h; } f;
    const _Float16* p = W + (size_t)col * Kpad + kbase + khalf * 8;
    f.u[0] = *(const uint4*)(p);
    f.u[1] = *(const uint4*)(p + 16);
    return f.h;
}

__device__ __forceinline__ float softplus_f(float v) {
    return (v > 20.0f) ? v : log1pf(__expf(v));
}

__global__ void wmma_gemm(const _Float16* __restrict__ A,
                          const _Float16* __restrict__ W,
                          const float* __restrict__ bias,
                          float* __restrict__ out,        // may be null
                          _Float16* __restrict__ out_h,   // may be null
                          int M, int N, int Kpad,
                          int hK, int hKpad, int epilogue) {
    const int lane  = threadIdx.x & 31;
    const int wave  = threadIdx.x >> 5;
    const int mtile = blockIdx.y * 8 + wave;       // 8 waves per block over M
    const int ntile = blockIdx.x;
    if (mtile * 16 >= M) return;                   // wave-uniform exit

    const int khalf = lane >> 4;
    const int rowA  = mtile * 16 + (lane & 15);
    const int colB  = ntile * 16 + (lane & 15);

    v8f acc = {};
    for (int kb = 0; kb < Kpad; kb += 32) {
        v16h a = load_fragA(A, rowA, Kpad, kb, khalf);
        v16h b = load_fragB(W, colB, Kpad, kb, khalf);
        acc = __builtin_amdgcn_wmma_f32_16x16x32_f16(
            /*neg_a=*/false, a, /*neg_b=*/false, b,
            /*c_mod=*/(short)0, acc, /*reuse_a=*/false, /*reuse_b=*/false);
    }

    const int col   = ntile * 16 + (lane & 15);
    const int rbase = mtile * 16 + (lane >> 4) * 8;
    const float bv  = (epilogue != EP_NONE && bias) ? bias[col] : 0.0f;
    #pragma unroll
    for (int v = 0; v < 8; ++v) {
        float r = acc[v] + bv;
        if (epilogue == EP_BIAS_SOFTPLUS) r = softplus_f(r);
        const int row = rbase + v;
        if (out) out[(size_t)row * N + col] = r;
        if (out_h) {
            if (col < hK)
                out_h[(size_t)row * hKpad + col] = (_Float16)r;
            else if (col < hKpad)
                out_h[(size_t)row * hKpad + col] = (_Float16)0.0f;
        }
    }
}

// ---------------------------------------------------------------------------
// Depthwise causal conv(4) + bias + SiLU over L within each batch.
// xz layout: [ROWS, 2*D_INNER]; xc = cols [0,512), z = cols [512,1024)
// Writes f32 (for the scan) and f16 (for the x-projection WMMA GEMM).
// ---------------------------------------------------------------------------
__global__ void conv_silu_kernel(const float* __restrict__ xz,
                                 const float* __restrict__ cw,
                                 const float* __restrict__ cb,
                                 float* __restrict__ xconv,
                                 _Float16* __restrict__ xconv_h) {
    int i = blockIdx.x * blockDim.x + threadIdx.x;
    if (i >= ROWS * D_INNER) return;
    int r = i / D_INNER;
    int d = i - r * D_INNER;
    int b = r / LSEQ;
    int l = r - b * LSEQ;
    float s = cb[d];
    #pragma unroll
    for (int k = 0; k < D_CONV; ++k) {
        int lk = l - (D_CONV - 1) + k;
        if (lk >= 0)
            s = __fmaf_rn(cw[d * D_CONV + k],
                          xz[(size_t)(b * LSEQ + lk) * (2 * D_INNER) + d], s);
    }
    float v = s / (1.0f + __expf(-s));  // SiLU
    xconv[i]   = v;
    xconv_h[i] = (_Float16)v;
}

// ---------------------------------------------------------------------------
// Selective scan: one wave32 per (b, d) channel (1024 waves).
// Each lane holds 4 of the 128 states in registers; y = sum_n h*C via
// shfl_xor butterfly.  Epilogue fuses +D*x and *silu(z) and writes the f16
// input of the out-projection WMMA GEMM directly.
// ---------------------------------------------------------------------------
__global__ void scan_kernel(const float* __restrict__ dtf,    // [ROWS, 512]
                            const float* __restrict__ dbl,    // [ROWS, 272]
                            const float* __restrict__ xconv,  // [ROWS, 512]
                            const float* __restrict__ xz,     // [ROWS, 1024]
                            const float* __restrict__ A_log,  // [512, 128]
                            const float* __restrict__ Dp,     // [512]
                            _Float16* __restrict__ y_h) {     // [ROWS, 512]
    const int lane = threadIdx.x & 31;
    const int wave = (blockIdx.x * blockDim.x + threadIdx.x) >> 5;
    if (wave >= BATCH * D_INNER) return;
    const int b = wave >> 9;          // /512
    const int d = wave & (D_INNER - 1);

    float a[4], h[4];
    #pragma unroll
    for (int j = 0; j < 4; ++j) {
        a[j] = -__expf(A_log[(size_t)d * D_STATE + lane + 32 * j]);
        h[j] = 0.0f;
    }
    const float Dv = Dp[d];

    for (int l = 0; l < LSEQ; ++l) {
        const int r = b * LSEQ + l;
        const float dtv = dtf[(size_t)r * D_INNER + d];
        const float xv  = xconv[(size_t)r * D_INNER + d];
        const float* row = dbl + (size_t)r * DBL;
        if (l + 1 < LSEQ)  // pull next timestep's B/C row (global_prefetch_b8)
            __builtin_prefetch(dbl + (size_t)(r + 1) * DBL + lane * 8, 0, 0);

        const float dtx = dtv * xv;
        float part = 0.0f;
        #pragma unroll
        for (int j = 0; j < 4; ++j) {
            const float Bv = row[DT_RANK + lane + 32 * j];
            const float Cv = row[DT_RANK + D_STATE + lane + 32 * j];
            h[j] = __fmaf_rn(__expf(dtv * a[j]), h[j], dtx * Bv);
            part = __fmaf_rn(h[j], Cv, part);
        }
        #pragma unroll
        for (int off = 16; off > 0; off >>= 1)
            part += __shfl_xor(part, off, 32);

        if (lane == 0) {
            const float zv = xz[(size_t)r * (2 * D_INNER) + D_INNER + d];
            const float yv = part + Dv * xv;
            y_h[(size_t)r * D_INNER + d] = (_Float16)(yv * (zv / (1.0f + __expf(-zv))));
        }
    }
}

// ---------------------------------------------------------------------------
// Final mean over L: (1248, 256) -> (2, 256)
// ---------------------------------------------------------------------------
__global__ void mean_kernel(const float* __restrict__ emb, float* __restrict__ out) {
    int i = blockIdx.x * blockDim.x + threadIdx.x;
    if (i >= BATCH * EMB) return;
    int b = i >> 8;
    int e = i & (EMB - 1);
    float s = 0.0f;
    for (int l = 0; l < LSEQ; ++l)
        s += emb[(size_t)(b * LSEQ + l) * EMB + e];
    out[i] = s * (1.0f / (float)LSEQ);
}

// ---------------------------------------------------------------------------
// Host side
// ---------------------------------------------------------------------------
static inline int cdiv(int a, int b) { return (a + b - 1) / b; }

extern "C" void kernel_launch(void* const* d_in, const int* in_sizes, int n_in,
                              void* d_out, int out_size, void* d_ws, size_t ws_size,
                              hipStream_t stream) {
    (void)in_sizes; (void)n_in; (void)out_size; (void)ws_size;

    const float* x      = (const float*)d_in[0];
    const float* proj_w = (const float*)d_in[1];
    const float* proj_b = (const float*)d_in[2];

    // Bump allocator over the workspace (256-byte aligned)
    char* ws = (char*)d_ws;
    auto alloc = [&](size_t bytes) -> void* {
        void* p = (void*)ws;
        ws += (bytes + 255) & ~(size_t)255;
        return p;
    };

    // Activations
    _Float16* spec_h  = (_Float16*)alloc((size_t)ROWS * KF_PAD    * 2);
    float*    emb     = (float*)   alloc((size_t)ROWS * EMB       * 4);
    _Float16* emb_h   = (_Float16*)alloc((size_t)ROWS * EMB       * 2);
    float*    xz      = (float*)   alloc((size_t)ROWS * 2*D_INNER * 4);
    float*    xconv   = (float*)   alloc((size_t)ROWS * D_INNER   * 4);
    _Float16* xconv_h = (_Float16*)alloc((size_t)ROWS * D_INNER   * 2);
    float*    dblb    = (float*)   alloc((size_t)ROWS * DBL       * 4);
    _Float16* dt_h    = (_Float16*)alloc((size_t)ROWS * DT_PAD    * 2);
    float*    dtf     = (float*)   alloc((size_t)ROWS * D_INNER   * 4);
    _Float16* y_h     = (_Float16*)alloc((size_t)ROWS * D_INNER   * 2);
    // Per-layer staged f16 weights (all cast in one launch)
    _Float16* proj_wh = (_Float16*)alloc((size_t)EMB * KF_PAD * 2);
    _Float16* in_wh[4]; _Float16* x_wh[4]; _Float16* dt_wh[4]; _Float16* out_wh[4];
    for (int L = 0; L < 4; ++L) {
        in_wh[L]  = (_Float16*)alloc((size_t)(2*D_INNER) * EMB * 2);
        x_wh[L]   = (_Float16*)alloc((size_t)DBL * D_INNER * 2);
        dt_wh[L]  = (_Float16*)alloc((size_t)D_INNER * DT_PAD * 2);
        out_wh[L] = (_Float16*)alloc((size_t)EMB * D_INNER * 2);
    }

    const int TB = 256;

    // ---- 1) STFT magnitude (padded f16 output) ----
    stft_kernel<<<cdiv(ROWS * KF_PAD, TB), TB, 0, stream>>>(x, spec_h);

    // ---- 2) Stage all 17 weight matrices to f16 in one launch ----
    CastJobs J;
    int jn = 0, acc = 0;
    auto addJob = [&](const float* s, _Float16* dmat, int srcStride, int K, int Kpad, int rows) {
        J.src[jn] = s; J.dst[jn] = dmat;
        J.srcStride[jn] = srcStride; J.K[jn] = K; J.Kpad[jn] = Kpad;
        J.off[jn] = acc; acc += rows * Kpad; ++jn;
    };
    addJob(proj_w, proj_wh, N_FREQ, N_FREQ, KF_PAD, EMB);
    for (int L = 0; L < 4; ++L) {
        const int pb = 3 + 9 * L;
        addJob((const float*)d_in[pb + 0], in_wh[L],  EMB,     EMB,     EMB,     2 * D_INNER);
        addJob((const float*)d_in[pb + 3], x_wh[L],   D_INNER, D_INNER, D_INNER, DBL);
        addJob((const float*)d_in[pb + 4], dt_wh[L],  DT_RANK, DT_RANK, DT_PAD,  D_INNER);
        addJob((const float*)d_in[pb + 8], out_wh[L], D_INNER, D_INNER, D_INNER, EMB);
    }
    J.off[NW] = acc;
    cast_weights_kernel<<<cdiv(acc, TB), TB, 0, stream>>>(J);

    auto gemm = [&](const _Float16* A, const _Float16* W, const float* bias,
                    float* outp, _Float16* outh, int M, int N, int Kpad,
                    int hK, int hKpad, int ep) {
        dim3 grid(N / 16, cdiv(M / 16, 8));
        wmma_gemm<<<grid, TB, 0, stream>>>(A, W, bias, outp, outh, M, N, Kpad, hK, hKpad, ep);
    };

    // ---- 3) Frequency projection: emb(+emb_h) = spec @ proj_w^T + proj_b ----
    gemm(spec_h, proj_wh, proj_b, emb, emb_h, ROWS, EMB, KF_PAD, EMB, EMB, EP_BIAS);

    // ---- 4) Four Mamba layers ----
    for (int L = 0; L < 4; ++L) {
        const int pb = 3 + 9 * L;
        const float* conv_w = (const float*)d_in[pb + 1];  // (512, 4)
        const float* conv_b = (const float*)d_in[pb + 2];  // (512,)
        const float* dt_b   = (const float*)d_in[pb + 5];  // (512,)
        const float* A_log  = (const float*)d_in[pb + 6];  // (512, 128)
        const float* Dp     = (const float*)d_in[pb + 7];  // (512,)

        // in-projection: xz = emb @ in_w^T   (N=1024, K=256)
        gemm(emb_h, in_wh[L], nullptr, xz, nullptr, ROWS, 2 * D_INNER, EMB, 0, 0, EP_NONE);

        // depthwise causal conv + bias + SiLU (writes f32 + f16)
        conv_silu_kernel<<<cdiv(ROWS * D_INNER, TB), TB, 0, stream>>>(
            xz, conv_w, conv_b, xconv, xconv_h);

        // x-projection: dbl = xconv @ x_w^T  (N=272, K=512)
        // fused f16 output: dt_h = f16(dbl[:, :16]) zero-padded to 32 cols
        gemm(xconv_h, x_wh[L], nullptr, dblb, dt_h, ROWS, DBL, D_INNER,
             DT_RANK, DT_PAD, EP_NONE);

        // dt projection + softplus: dtf = softplus(dt_h @ dt_w^T + dt_b)
        gemm(dt_h, dt_wh[L], dt_b, dtf, nullptr, ROWS, D_INNER, DT_PAD, 0, 0,
             EP_BIAS_SOFTPLUS);

        // selective scan (fused +D*x and *silu(z)); writes f16 y directly
        scan_kernel<<<(BATCH * D_INNER * 32) / TB, TB, 0, stream>>>(
            dtf, dblb, xconv, xz, A_log, Dp, y_h);

        // out-projection: emb(+emb_h for next layer) = y @ out_w^T (N=256, K=512)
        gemm(y_h, out_wh[L], nullptr, emb, emb_h, ROWS, EMB, D_INNER, EMB, EMB, EP_NONE);
    }

    // ---- 5) Mean pool -> (2, 256) ----
    mean_kernel<<<cdiv(BATCH * EMB, TB), TB, 0, stream>>>(emb, (float*)d_out);
}
